// AttentionWithSpatial_6305011990580
// MI455X (gfx1250) — matrix operand order repaired and checked by reading, not compiled, not run on hardware
//
#include <hip/hip_runtime.h>
#include <hip/hip_bf16.h>

#define B_  4
#define H_  4
#define N_  2048
#define D_  64
#define DIM_ 256
#define SCALE_ 0.125f

typedef __attribute__((ext_vector_type(16))) __bf16 bf16x16;
typedef __attribute__((ext_vector_type(8)))  float  f32x8;

struct __attribute__((aligned(16))) U4 { unsigned int x, y, z, w; };
struct __attribute__((aligned(16))) F4 { float x, y, z, w; };

union ABu { unsigned int u[8]; bf16x16 v; };

// Truncating f32->bf16 (RTZ): single shift; backend can fold into d16_hi stores.
__device__ inline unsigned short f32_to_bf16(float f) {
    return (unsigned short)(__float_as_uint(f) >> 16);
}
// Pack bf16(a) | bf16(b)<<16 in ONE v_perm_b32:
//   result byte0,1 = a bytes 2,3 ; result byte2,3 = b bytes 2,3
__device__ inline unsigned int pack2bf(float a, float b) {
    return __builtin_amdgcn_perm(__float_as_uint(b), __float_as_uint(a), 0x07060302u);
}
__device__ inline f32x8 zero8() {
    f32x8 z;
#pragma unroll
    for (int i = 0; i < 8; ++i) z[i] = 0.f;
    return z;
}
__device__ inline f32x8 wmma_bf16(const ABu& a, const ABu& b, f32x8 c) {
    return __builtin_amdgcn_wmma_f32_16x16x32_bf16(
        /*neg_a=*/false, a.v, /*neg_b=*/false, b.v,
        /*c_mod=*/(short)0, c, /*reuse_a=*/false, /*reuse_b=*/false);
}

// -------------------------------------------------------------------------
// Kernel 1: QKV projection.  x[8192,256]f32 @ Wqkv[256,768]f32 -> bf16
//   q,k stored [hb][n][64]; v stored transposed [hb][64][n].
// WG = 256 threads = 8 waves: 2 M-tiles x 4 N-tiles (32 rows x 64 cols).
// -------------------------------------------------------------------------
__global__ __launch_bounds__(256) void qkv_gemm_kernel(
    const float* __restrict__ x, const float* __restrict__ wqkv,
    unsigned short* __restrict__ qg, unsigned short* __restrict__ kg,
    unsigned short* __restrict__ vtg)
{
    __shared__ __attribute__((aligned(16))) unsigned short wt[64][264]; // [col][k], pitch 528B

    const int nblk = blockIdx.x % 12;
    const int mblk = blockIdx.x / 12;
    const int n0 = nblk * 64;
    const int wave = threadIdx.x >> 5, lane = threadIdx.x & 31;
    const int ln = lane & 15, lh = lane >> 4;
    const int mt = wave >> 2, ntile = wave & 3;
    const int m0 = mblk * 32 + mt * 16;

    // Cooperative load of W tile, transposed into LDS as bf16.
    {
        int c = threadIdx.x & 63;
        int kb0 = (threadIdx.x >> 6) * 64;
        for (int kk = 0; kk < 64; ++kk) {
            int k = kb0 + kk;
            wt[c][k] = f32_to_bf16(wqkv[(size_t)k * 768 + n0 + c]);
        }
    }
    __syncthreads();

    f32x8 acc = zero8();
#pragma unroll
    for (int kb = 0; kb < 8; ++kb) {
        ABu a;
        const float* xp = x + (size_t)(m0 + ln) * DIM_ + kb * 32 + lh * 8;
        F4 f0 = *(const F4*)(xp);
        F4 f1 = *(const F4*)(xp + 4);
        F4 f2 = *(const F4*)(xp + 16);
        F4 f3 = *(const F4*)(xp + 20);
        a.u[0] = pack2bf(f0.x, f0.y); a.u[1] = pack2bf(f0.z, f0.w);
        a.u[2] = pack2bf(f1.x, f1.y); a.u[3] = pack2bf(f1.z, f1.w);
        a.u[4] = pack2bf(f2.x, f2.y); a.u[5] = pack2bf(f2.z, f2.w);
        a.u[6] = pack2bf(f3.x, f3.y); a.u[7] = pack2bf(f3.z, f3.w);

        ABu bm;
        const unsigned short* wp = &wt[ntile * 16 + ln][kb * 32 + lh * 8];
        U4 b0 = *(const U4*)(wp);
        U4 b1 = *(const U4*)(wp + 16);
        bm.u[0] = b0.x; bm.u[1] = b0.y; bm.u[2] = b0.z; bm.u[3] = b0.w;
        bm.u[4] = b1.x; bm.u[5] = b1.y; bm.u[6] = b1.z; bm.u[7] = b1.w;

        acc = wmma_bf16(a, bm, acc);
    }

    const int c = n0 + ntile * 16 + ln;
    const int sect = c >> 8;     // 0=q 1=k 2=v
    const int cc = c & 255;
    const int hh = cc >> 6, dd = cc & 63;
#pragma unroll
    for (int r = 0; r < 8; ++r) {
        int row_f = m0 + r + 8 * lh;
        int bb = row_f >> 11, ii = row_f & (N_ - 1);
        unsigned short bv = f32_to_bf16(acc[r]);
        size_t hb2 = (size_t)(hh * B_ + bb);
        if (sect == 0)      qg[(hb2 * N_ + ii) * D_ + dd] = bv;
        else if (sect == 1) kg[(hb2 * N_ + ii) * D_ + dd] = bv;
        else                vtg[(hb2 * D_ + dd) * N_ + ii] = bv;
    }
}

// -------------------------------------------------------------------------
// Kernel 2: flash attention with spatial bias + mask.
// grid = (H*B) * (N/128); WG = 8 waves, each wave owns a 16-row query tile.
// -------------------------------------------------------------------------
__global__ __launch_bounds__(256) void flash_attn_kernel(
    const unsigned short* __restrict__ qg, const unsigned short* __restrict__ kg,
    const unsigned short* __restrict__ vtg,
    const float* __restrict__ spatial, const int* __restrict__ mask,
    unsigned short* __restrict__ attn_out)
{
    __shared__ __attribute__((aligned(16))) unsigned short plds[8][16][40]; // 80B pitch

    const int hb = blockIdx.x >> 4;
    const int qb = blockIdx.x & 15;
    const int h = hb >> 2, b = hb & 3;
    const int wave = threadIdx.x >> 5, lane = threadIdx.x & 31;
    const int ln = lane & 15, lh = lane >> 4;
    const int qrow0 = qb * 128 + wave * 16;

    const unsigned short* kg_hb = kg + (size_t)hb * N_ * D_;
    const unsigned short* vt_hb = vtg + (size_t)hb * D_ * N_;
    const float* sp_b = spatial + (size_t)b * N_ * N_;
    const int* mk_b = mask + (size_t)b * N_ * N_;

    // Q A-fragments for d-blocks 0..31, 32..63: resident across the key loop.
    ABu qa[2];
    {
        const unsigned short* qp = qg + ((size_t)hb * N_ + qrow0 + ln) * D_;
#pragma unroll
        for (int kb = 0; kb < 2; ++kb) {
            const unsigned short* p = qp + kb * 32 + lh * 8;
            U4 t0 = *(const U4*)(p);
            U4 t1 = *(const U4*)(p + 16);
            qa[kb].u[0] = t0.x; qa[kb].u[1] = t0.y; qa[kb].u[2] = t0.z; qa[kb].u[3] = t0.w;
            qa[kb].u[4] = t1.x; qa[kb].u[5] = t1.y; qa[kb].u[6] = t1.z; qa[kb].u[7] = t1.w;
        }
    }

    f32x8 o[4];
#pragma unroll
    for (int nt = 0; nt < 4; ++nt) o[nt] = zero8();
    float m[8], l[8];
#pragma unroll
    for (int r = 0; r < 8; ++r) { m[r] = -3e38f; l[r] = 0.f; }

    for (int j = 0; j < N_; j += 32) {
        float sbuf[2][8];
        float tilemax[8];
#pragma unroll
        for (int r = 0; r < 8; ++r) tilemax[r] = -3e38f;

#pragma unroll
        for (int t = 0; t < 2; ++t) {
            const int jt = j + t * 16;
            // K B-fragments (B^T[N=keycol][K=d] == k row-major): contiguous.
            ABu kbf[2];
            const unsigned short* kp = kg_hb + (size_t)(jt + ln) * D_ + lh * 8;
#pragma unroll
            for (int kb = 0; kb < 2; ++kb) {
                U4 b0 = *(const U4*)(kp + kb * 32);
                U4 b1 = *(const U4*)(kp + kb * 32 + 16);
                kbf[kb].u[0] = b0.x; kbf[kb].u[1] = b0.y; kbf[kb].u[2] = b0.z; kbf[kb].u[3] = b0.w;
                kbf[kb].u[4] = b1.x; kbf[kb].u[5] = b1.y; kbf[kb].u[6] = b1.z; kbf[kb].u[7] = b1.w;
            }
            f32x8 s = zero8();
            s = wmma_bf16(qa[0], kbf[0], s);
            s = wmma_bf16(qa[1], kbf[1], s);

            const int col = jt + ln;
#pragma unroll
            for (int r = 0; r < 8; ++r) {
                int row = qrow0 + r + 8 * lh;
                size_t idx = (size_t)row * N_ + col;
                float sv = s[r] * SCALE_ + sp_b[idx];
                sv = (mk_b[idx] == 0) ? -3e38f : sv;
                sbuf[t][r] = sv;
                float v = sv;
                v = fmaxf(v, __shfl_xor(v, 1, 32));
                v = fmaxf(v, __shfl_xor(v, 2, 32));
                v = fmaxf(v, __shfl_xor(v, 4, 32));
                v = fmaxf(v, __shfl_xor(v, 8, 32));
                tilemax[r] = fmaxf(tilemax[r], v);
            }
        }

        // Online softmax update.
        float corr[8];
#pragma unroll
        for (int r = 0; r < 8; ++r) {
            float nm = fmaxf(m[r], tilemax[r]);
            corr[r] = __expf(m[r] - nm);
            m[r] = nm;
        }
#pragma unroll
        for (int r = 0; r < 8; ++r) {
            float p0 = __expf(sbuf[0][r] - m[r]);
            float p1 = __expf(sbuf[1][r] - m[r]);
            plds[wave][r + 8 * lh][ln]      = f32_to_bf16(p0);
            plds[wave][r + 8 * lh][16 + ln] = f32_to_bf16(p1);
            float rs = p0 + p1;
            rs += __shfl_xor(rs, 1, 32);
            rs += __shfl_xor(rs, 2, 32);
            rs += __shfl_xor(rs, 4, 32);
            rs += __shfl_xor(rs, 8, 32);
            l[r] = l[r] * corr[r] + rs;
        }
#pragma unroll
        for (int nt = 0; nt < 4; ++nt)
#pragma unroll
            for (int r = 0; r < 8; ++r) o[nt][r] *= corr[r];

        // Re-tile P (C layout -> A layout) through this wave's private LDS slab.
        ABu pA;
        {
            const unsigned short* pr = &plds[wave][ln][lh * 8];
            U4 t0 = *(const U4*)(pr);
            U4 t1 = *(const U4*)(pr + 16);
            pA.u[0] = t0.x; pA.u[1] = t0.y; pA.u[2] = t0.z; pA.u[3] = t0.w;
            pA.u[4] = t1.x; pA.u[5] = t1.y; pA.u[6] = t1.z; pA.u[7] = t1.w;
        }
        // P @ V with V^T in global memory: per-lane contiguous B fragments.
#pragma unroll
        for (int nt = 0; nt < 4; ++nt) {
            ABu vB;
            const unsigned short* vp = vt_hb + (size_t)(nt * 16 + ln) * N_ + j + lh * 8;
            U4 b0 = *(const U4*)(vp);
            U4 b1 = *(const U4*)(vp + 16);
            vB.u[0] = b0.x; vB.u[1] = b0.y; vB.u[2] = b0.z; vB.u[3] = b0.w;
            vB.u[4] = b1.x; vB.u[5] = b1.y; vB.u[6] = b1.z; vB.u[7] = b1.w;
            o[nt] = wmma_bf16(pA, vB, o[nt]);
        }

        if (j + 32 < N_) {
            __builtin_prefetch(kg_hb + (size_t)(j + 32 + ln) * D_ + lh * 16, 0, 1);
            __builtin_prefetch(vt_hb + (size_t)ln * N_ + j + 32 + lh * 16, 0, 1);
        }
    }

    // Normalize and store bf16 [b][n][h*64+d] for the output projection.
#pragma unroll
    for (int nt = 0; nt < 4; ++nt) {
#pragma unroll
        for (int r = 0; r < 8; ++r) {
            float val = o[nt][r] / l[r];
            int row = qrow0 + r + 8 * lh;
            int colc = h * 64 + nt * 16 + ln;
            attn_out[((size_t)b * N_ + row) * DIM_ + colc] = f32_to_bf16(val);
        }
    }
}

// -------------------------------------------------------------------------
// Kernel 3: output projection attn[8192,256]bf16 @ Wout[256,256] + bias -> f32
// -------------------------------------------------------------------------
__global__ __launch_bounds__(256) void out_gemm_kernel(
    const unsigned short* __restrict__ attn, const float* __restrict__ wout,
    const float* __restrict__ bout, float* __restrict__ out)
{
    __shared__ __attribute__((aligned(16))) unsigned short wt[64][264];

    const int nblk = blockIdx.x % 4;
    const int mblk = blockIdx.x / 4;
    const int n0 = nblk * 64;
    const int wave = threadIdx.x >> 5, lane = threadIdx.x & 31;
    const int ln = lane & 15, lh = lane >> 4;
    const int mt = wave >> 2, ntile = wave & 3;
    const int m0 = mblk * 32 + mt * 16;

    {
        int c = threadIdx.x & 63;
        int kb0 = (threadIdx.x >> 6) * 64;
        for (int kk = 0; kk < 64; ++kk) {
            int k = kb0 + kk;
            wt[c][k] = f32_to_bf16(wout[(size_t)k * DIM_ + n0 + c]);
        }
    }
    __syncthreads();

    f32x8 acc = zero8();
#pragma unroll
    for (int kb = 0; kb < 8; ++kb) {
        ABu a;
        const unsigned short* ap = attn + (size_t)(m0 + ln) * DIM_ + kb * 32 + lh * 8;
        U4 t0 = *(const U4*)(ap);
        U4 t1 = *(const U4*)(ap + 16);
        a.u[0] = t0.x; a.u[1] = t0.y; a.u[2] = t0.z; a.u[3] = t0.w;
        a.u[4] = t1.x; a.u[5] = t1.y; a.u[6] = t1.z; a.u[7] = t1.w;

        ABu bm;
        const unsigned short* wp = &wt[ntile * 16 + ln][kb * 32 + lh * 8];
        U4 b0 = *(const U4*)(wp);
        U4 b1 = *(const U4*)(wp + 16);
        bm.u[0] = b0.x; bm.u[1] = b0.y; bm.u[2] = b0.z; bm.u[3] = b0.w;
        bm.u[4] = b1.x; bm.u[5] = b1.y; bm.u[6] = b1.z; bm.u[7] = b1.w;

        acc = wmma_bf16(a, bm, acc);
    }

    const int c = n0 + ntile * 16 + ln;
    const float bias = bout[c];
#pragma unroll
    for (int r = 0; r < 8; ++r) {
        int row_f = m0 + r + 8 * lh;
        out[(size_t)row_f * DIM_ + c] = acc[r] + bias;
    }
}

// -------------------------------------------------------------------------
extern "C" void kernel_launch(void* const* d_in, const int* in_sizes, int n_in,
                              void* d_out, int out_size, void* d_ws, size_t ws_size,
                              hipStream_t stream) {
    const float* x       = (const float*)d_in[0];
    const int*   mask    = (const int*)d_in[1];
    const float* spatial = (const float*)d_in[2];
    const float* wqkv    = (const float*)d_in[3];
    const float* wout    = (const float*)d_in[4];
    const float* bout    = (const float*)d_in[5];
    float* out = (float*)d_out;

    const size_t qkv_elems = (size_t)H_ * B_ * N_ * D_;   // 2,097,152
    unsigned short* qg   = (unsigned short*)d_ws;
    unsigned short* kg   = qg + qkv_elems;
    unsigned short* vtg  = kg + qkv_elems;
    unsigned short* attn = vtg + qkv_elems;

    // 1) QKV projection: 256 M-blocks x 12 N-blocks
    qkv_gemm_kernel<<<dim3(256 * 12), dim3(256), 0, stream>>>(x, wqkv, qg, kg, vtg);
    // 2) Flash attention: 16 head-batch x 16 query-blocks
    flash_attn_kernel<<<dim3(256), dim3(256), 0, stream>>>(qg, kg, vtg, spatial, mask, attn);
    // 3) Output projection: 256 M-blocks x 4 N-blocks
    out_gemm_kernel<<<dim3(256 * 4), dim3(256), 0, stream>>>(attn, wout, bout, out);
}
